// Adapter_56246891709114
// MI455X (gfx1250) — matrix operand order repaired and checked by reading, not compiled, not run on hardware
//
#include <hip/hip_runtime.h>
#include <hip/hip_bf16.h>
#include <math.h>

// ---------------------------------------------------------------------------
// CDNA5 (gfx1250) wave32 WMMA bf16 path.
//   v_wmma_f32_16x16x32_bf16 : A 16x32 bf16, B 32x16 bf16, C/D 16x16 f32
// A layout (ISA 7.12.2, 16-bit A 16x32):
//   lane l: row = l&15 ; half = l>>4 ; kb = half*8
//   elem j(0..7)  -> K = kb + j
//   elem j(8..15) -> K = 16 + kb + (j-8)
// B layout (K x N): lane l: col = l&15 ; elem j -> K = (j&8)*2 + (l>>4)*8 + (j&7) (same order as A)
// C/D layout: lane l holds D[(l>>4)*8 + r][l&15], r = VGPR 0..7
// ---------------------------------------------------------------------------

typedef __attribute__((ext_vector_type(16))) __bf16 v16bf;
typedef __attribute__((ext_vector_type(8)))  float  v8f;

union Frag {
    v16bf v;
    uint4 q[2];
};

// hardware fptrunc (RNE): lowers to v_cvt_pk_bf16_f32 pairs
static __device__ inline __bf16 tobf(float f) { return (__bf16)f; }

// A-frag from one fp32 row (elements p[kb..kb+7] and p[16+kb..16+kb+7])
static __device__ inline void frag_f32(const float* p, int kb, Frag& f) {
    const float4* P0 = (const float4*)(p + kb);
    const float4* P1 = (const float4*)(p + 16 + kb);
    float4 a0 = P0[0], a1 = P0[1], b0 = P1[0], b1 = P1[1];
    f.v[0]  = tobf(a0.x); f.v[1]  = tobf(a0.y); f.v[2]  = tobf(a0.z); f.v[3]  = tobf(a0.w);
    f.v[4]  = tobf(a1.x); f.v[5]  = tobf(a1.y); f.v[6]  = tobf(a1.z); f.v[7]  = tobf(a1.w);
    f.v[8]  = tobf(b0.x); f.v[9]  = tobf(b0.y); f.v[10] = tobf(b0.z); f.v[11] = tobf(b0.w);
    f.v[12] = tobf(b1.x); f.v[13] = tobf(b1.y); f.v[14] = tobf(b1.z); f.v[15] = tobf(b1.w);
}

// A-frag from sum of two fp32 rows
static __device__ inline void frag_f32_sum(const float* p, const float* q, int kb, Frag& f) {
    const float4* P0 = (const float4*)(p + kb);
    const float4* Q0 = (const float4*)(q + kb);
    const float4* P1 = (const float4*)(p + 16 + kb);
    const float4* Q1 = (const float4*)(q + 16 + kb);
    float4 a0 = P0[0], c0 = Q0[0], a1 = P0[1], c1 = Q0[1];
    float4 b0 = P1[0], d0 = Q1[0], b1 = P1[1], d1 = Q1[1];
    f.v[0]  = tobf(a0.x + c0.x); f.v[1]  = tobf(a0.y + c0.y);
    f.v[2]  = tobf(a0.z + c0.z); f.v[3]  = tobf(a0.w + c0.w);
    f.v[4]  = tobf(a1.x + c1.x); f.v[5]  = tobf(a1.y + c1.y);
    f.v[6]  = tobf(a1.z + c1.z); f.v[7]  = tobf(a1.w + c1.w);
    f.v[8]  = tobf(b0.x + d0.x); f.v[9]  = tobf(b0.y + d0.y);
    f.v[10] = tobf(b0.z + d0.z); f.v[11] = tobf(b0.w + d0.w);
    f.v[12] = tobf(b1.x + d1.x); f.v[13] = tobf(b1.y + d1.y);
    f.v[14] = tobf(b1.z + d1.z); f.v[15] = tobf(b1.w + d1.w);
}

static __device__ inline v8f wmma_bf16(const Frag& A, const Frag& B, v8f C) {
    return __builtin_amdgcn_wmma_f32_16x16x32_bf16(
        /*neg_a=*/false, A.v, /*neg_b=*/false, B.v,
        /*c_mod=*/(short)0, C, /*reuse_a=*/false, /*reuse_b=*/false);
}

// ---------------------------------------------------------------------------
// Kernel 0: zero fill
// ---------------------------------------------------------------------------
__global__ void k_fill_zero(float* p, size_t n) {
    size_t i = (size_t)blockIdx.x * blockDim.x + threadIdx.x;
    if (i < n) p[i] = 0.0f;
}

// ---------------------------------------------------------------------------
// Kernel 1: pack fp32 weight [Kdim][Ncols] -> bf16 B-fragment layout
//   out[((t*chunks + c)*32 + l)*16 + j] = W[c*32 + (j&8)*2 + (l>>4)*8 + (j&7)][t*16 + (l&15)]
// ---------------------------------------------------------------------------
__global__ void k_pack_w(const float* __restrict__ W, int Kdim, int Ncols,
                         __bf16* __restrict__ out) {
    int idx = blockIdx.x * blockDim.x + threadIdx.x;
    int total = Kdim * Ncols;
    if (idx >= total) return;
    int j  = idx & 15;
    int l  = (idx >> 4) & 31;
    int ct = idx >> 9;
    int chunks = Kdim >> 5;
    int c = ct % chunks;
    int t = ct / chunks;
    int krow = c * 32 + (j & 8) * 2 + (l >> 4) * 8 + (j & 7);
    int col  = t * 16 + (l & 15);
    out[idx] = tobf(W[krow * Ncols + col]);
}

// ---------------------------------------------------------------------------
// Kernel 2: msg = relu(nfeat[src] + efeat); aggr[dst] += msg  (one wave/edge)
// ---------------------------------------------------------------------------
__global__ __launch_bounds__(256) void k_scatter(
    const float* __restrict__ nfeat, const float* __restrict__ efeat,
    const int* __restrict__ src, const int* __restrict__ dst,
    float* __restrict__ aggr, int E) {
    int e = blockIdx.x * (blockDim.x >> 5) + (threadIdx.x >> 5);
    if (e >= E) return;
    int lane = threadIdx.x & 31;
    int s = src[e], d = dst[e];
    float4 a = *((const float4*)(nfeat + (size_t)s * 128) + lane);
    float4 b = *((const float4*)(efeat + (size_t)e * 128) + lane);
    float m0 = fmaxf(a.x + b.x, 0.0f);
    float m1 = fmaxf(a.y + b.y, 0.0f);
    float m2 = fmaxf(a.z + b.z, 0.0f);
    float m3 = fmaxf(a.w + b.w, 0.0f);
    float* o = aggr + (size_t)d * 128 + lane * 4;
    atomicAdd(o + 0, m0);
    atomicAdd(o + 1, m1);
    atomicAdd(o + 2, m2);
    atomicAdd(o + 3, m3);
}

// ---------------------------------------------------------------------------
// Kernel 3: node pipeline. One wave handles a 16-row tile.
// ---------------------------------------------------------------------------
__global__ __launch_bounds__(128) void k_node(
    const float* __restrict__ nfeat, const float* __restrict__ aggr,
    const float* __restrict__ ndist,
    const __bf16* __restrict__ ginP, const float* __restrict__ gin_b,
    const __bf16* __restrict__ ndP,  const float* __restrict__ nd_b,
    const __bf16* __restrict__ nffnP, const float* __restrict__ nffn_b,
    const float* __restrict__ projW, const float* __restrict__ projb,
    __bf16* __restrict__ xg, float* __restrict__ out, int N) {

    __shared__ __bf16 sh_x[4][16 * 128];   // bf16 x tile, per wave
    __shared__ __bf16 sh_nd[4][16 * 32];   // bf16 nd tile, per wave
    __shared__ float  sh_h[4][16 * 128];   // f32 node_h tile, per wave

    const int wid  = threadIdx.x >> 5;
    const int lane = threadIdx.x & 31;
    const int half = lane >> 4;
    const int r16  = lane & 15;
    const int kb   = half * 8;

    const int tile = blockIdx.x * 4 + wid;
    const int m0   = tile * 16;
    if (m0 >= N) return;                 // uniform per-wave exit (no barriers used)
    const int row = m0 + r16;

    // ---- Phase 1: x = relu((nfeat+aggr) @ gin_W + b), K=128 (4 chunks)
    Frag a[5];
    {
        const float* pn = nfeat + (size_t)row * 128;
        const float* pa = aggr  + (size_t)row * 128;
#pragma unroll
        for (int c = 0; c < 4; ++c)
            frag_f32_sum(pn + c * 32, pa + c * 32, kb, a[c]);
    }
#pragma unroll
    for (int t = 0; t < 8; ++t) {
        v8f acc = {};
#pragma unroll
        for (int c = 0; c < 4; ++c) {
            Frag b;
            const __bf16* bp = ginP + ((size_t)(t * 4 + c) * 32 + lane) * 16;
            b.q[0] = *(const uint4*)(bp);
            b.q[1] = *(const uint4*)(bp + 8);
            acc = wmma_bf16(a[c], b, acc);
        }
        float bias = gin_b[t * 16 + r16];
#pragma unroll
        for (int r = 0; r < 8; ++r) {
            float v = fmaxf(acc[r] + bias, 0.0f);
            sh_x[wid][(half * 8 + r) * 128 + t * 16 + r16] = tobf(v);
        }
    }
    // coalesced store of bf16 x tile to global (for edge gathers; stays in L2)
    {
        const uint4* s = (const uint4*)sh_x[wid];
        uint4* g = (uint4*)(xg + (size_t)m0 * 128);
#pragma unroll
        for (int i = 0; i < 8; ++i) g[lane + 32 * i] = s[lane + 32 * i];
    }

    // ---- Phase 2: nd = relu(ndist @ ndW + b), K=32 (1 chunk), Nout=32 (2 tiles)
    {
        Frag an;
        frag_f32(ndist + (size_t)row * 32, kb, an);
#pragma unroll
        for (int t = 0; t < 2; ++t) {
            Frag b;
            const __bf16* bp = ndP + ((size_t)t * 32 + lane) * 16;
            b.q[0] = *(const uint4*)(bp);
            b.q[1] = *(const uint4*)(bp + 8);
            v8f acc = {};
            acc = wmma_bf16(an, b, acc);
            float bias = nd_b[t * 16 + r16];
#pragma unroll
            for (int r = 0; r < 8; ++r) {
                float v = fmaxf(acc[r] + bias, 0.0f);
                sh_nd[wid][(half * 8 + r) * 32 + t * 16 + r16] = tobf(v);
            }
        }
    }

    // ---- Phase 3: node_h = relu([x, nd] @ nffn_W + b), K=160 (5 chunks)
#pragma unroll
    for (int c = 0; c < 4; ++c) {
        a[c].q[0] = *(const uint4*)&sh_x[wid][r16 * 128 + c * 32 + kb];
        a[c].q[1] = *(const uint4*)&sh_x[wid][r16 * 128 + c * 32 + 16 + kb];
    }
    a[4].q[0] = *(const uint4*)&sh_nd[wid][r16 * 32 + kb];
    a[4].q[1] = *(const uint4*)&sh_nd[wid][r16 * 32 + 16 + kb];

#pragma unroll
    for (int t = 0; t < 8; ++t) {
        v8f acc = {};
#pragma unroll
        for (int c = 0; c < 5; ++c) {
            Frag b;
            const __bf16* bp = nffnP + ((size_t)(t * 5 + c) * 32 + lane) * 16;
            b.q[0] = *(const uint4*)(bp);
            b.q[1] = *(const uint4*)(bp + 8);
            acc = wmma_bf16(a[c], b, acc);
        }
        float bias = nffn_b[t * 16 + r16];
#pragma unroll
        for (int r = 0; r < 8; ++r)
            sh_h[wid][(half * 8 + r) * 128 + t * 16 + r16] = fmaxf(acc[r] + bias, 0.0f);
    }

    // ---- Phase 4: heads. lanes 0-15 -> col 0 (clip), lanes 16-31 -> col 1 (sigmoid)
    {
        float sum = projb[half];
        const float* H = &sh_h[wid][r16 * 128];
        const float2* PW = (const float2*)projW;   // [k].x = col0, [k].y = col1
        for (int k = 0; k < 128; ++k) {
            float2 w = PW[k];
            sum += H[k] * (half == 0 ? w.x : w.y);
        }
        int m = m0 + r16;
        if (half == 0) out[m]     = fminf(fmaxf(sum, 1.0f), 100.0f);
        else           out[N + m] = 1.0f / (1.0f + __expf(-sum));
    }
}

// ---------------------------------------------------------------------------
// Kernel 4: edge pipeline. One wave handles a 32-edge tile (2 x 16-row M tiles)
// so each B-fragment load feeds two WMMAs (halves per-edge weight traffic;
// the 72KB packed edge-FFN weight stream exceeds the 64KB WGP$).
//  pair   = relu([x[src], x[dst]])  (x >= 0 already, relu is identity)
//  ed     = relu(edist @ edW + edb)
//  edge_h = relu([pair, ed] @ effn_W + effn_b)  K=288 (9 chunks)
//  edge_o = edge_h @ projW + projb -> clip / sigmoid
// ---------------------------------------------------------------------------
__global__ __launch_bounds__(128) void k_edge(
    const __bf16* __restrict__ xg, const float* __restrict__ edist,
    const int* __restrict__ src, const int* __restrict__ dst,
    const __bf16* __restrict__ edP,  const float* __restrict__ ed_b,
    const __bf16* __restrict__ effnP, const float* __restrict__ effn_b,
    const float* __restrict__ projW, const float* __restrict__ projb,
    float* __restrict__ out, int E, int N) {

    __shared__ __bf16 sh_ed[4][2][16 * 32];    // 2KB / wave
    __shared__ __bf16 sh_h[4][2][16 * 128];    // 8KB / wave (bf16 staging for head)

    const int wid  = threadIdx.x >> 5;
    const int lane = threadIdx.x & 31;
    const int half = lane >> 4;
    const int r16  = lane & 15;
    const int kb   = half * 8;

    const int tile = blockIdx.x * 4 + wid;
    const int e0   = tile * 32;
    if (e0 >= E) return;                 // uniform per-wave exit (no barriers used)

    Frag a[2][9];

#pragma unroll
    for (int u = 0; u < 2; ++u) {
        const int e = e0 + u * 16 + r16;
        const int s = src[e];
        const int d = dst[e];

        // gather A-frags straight from bf16 x table (12.8 MB -> L2 resident)
        const __bf16* ps = xg + (size_t)s * 128;
        const __bf16* pd = xg + (size_t)d * 128;
#pragma unroll
        for (int c = 0; c < 4; ++c) {
            a[u][c].q[0]     = *(const uint4*)(ps + c * 32 + kb);
            a[u][c].q[1]     = *(const uint4*)(ps + c * 32 + 16 + kb);
            a[u][4 + c].q[0] = *(const uint4*)(pd + c * 32 + kb);
            a[u][4 + c].q[1] = *(const uint4*)(pd + c * 32 + 16 + kb);
        }

        // ed = relu(edist @ edW + b)
        Frag an;
        frag_f32(edist + (size_t)e * 32, kb, an);
#pragma unroll
        for (int t = 0; t < 2; ++t) {
            Frag b;
            const __bf16* bp = edP + ((size_t)t * 32 + lane) * 16;
            b.q[0] = *(const uint4*)(bp);
            b.q[1] = *(const uint4*)(bp + 8);
            v8f acc = {};
            acc = wmma_bf16(an, b, acc);
            float bias = ed_b[t * 16 + r16];
#pragma unroll
            for (int r = 0; r < 8; ++r) {
                float v = fmaxf(acc[r] + bias, 0.0f);
                sh_ed[wid][u][(half * 8 + r) * 32 + t * 16 + r16] = tobf(v);
            }
        }
        a[u][8].q[0] = *(const uint4*)&sh_ed[wid][u][r16 * 32 + kb];
        a[u][8].q[1] = *(const uint4*)&sh_ed[wid][u][r16 * 32 + 16 + kb];
    }

    // edge_h = relu([xs, xd, ed] @ effn_W + b): each B-frag feeds 2 WMMAs
#pragma unroll
    for (int t = 0; t < 8; ++t) {
        v8f acc0 = {}, acc1 = {};
#pragma unroll
        for (int c = 0; c < 9; ++c) {
            Frag b;
            const __bf16* bp = effnP + ((size_t)(t * 9 + c) * 32 + lane) * 16;
            b.q[0] = *(const uint4*)(bp);
            b.q[1] = *(const uint4*)(bp + 8);
            acc0 = wmma_bf16(a[0][c], b, acc0);
            acc1 = wmma_bf16(a[1][c], b, acc1);
        }
        float bias = effn_b[t * 16 + r16];
#pragma unroll
        for (int r = 0; r < 8; ++r) {
            sh_h[wid][0][(half * 8 + r) * 128 + t * 16 + r16] = tobf(fmaxf(acc0[r] + bias, 0.0f));
            sh_h[wid][1][(half * 8 + r) * 128 + t * 16 + r16] = tobf(fmaxf(acc1[r] + bias, 0.0f));
        }
    }

    // heads. lanes 0-15 -> col 0 (clip), lanes 16-31 -> col 1 (sigmoid)
    const float2* PW = (const float2*)projW;
#pragma unroll
    for (int u = 0; u < 2; ++u) {
        float sum = projb[half];
        const __bf16* H = &sh_h[wid][u][r16 * 128];
        for (int k = 0; k < 128; ++k) {
            float2 w = PW[k];
            sum += (float)H[k] * (half == 0 ? w.x : w.y);
        }
        int ei = e0 + u * 16 + r16;
        if (half == 0) out[2 * N + ei]     = fminf(fmaxf(sum, 1.0f), 100.0f);
        else           out[2 * N + E + ei] = 1.0f / (1.0f + __expf(-sum));
    }
}

// ---------------------------------------------------------------------------
// Host launcher
// ---------------------------------------------------------------------------
extern "C" void kernel_launch(void* const* d_in, const int* in_sizes, int n_in,
                              void* d_out, int out_size, void* d_ws, size_t ws_size,
                              hipStream_t stream) {
    const int N = in_sizes[0] / 128;   // 50000
    const int E = in_sizes[1] / 128;   // 800000

    const float* nfeat  = (const float*)d_in[0];
    const float* efeat  = (const float*)d_in[1];
    const float* ndist  = (const float*)d_in[2];
    const float* edist  = (const float*)d_in[3];
    const int*   src    = (const int*)d_in[4];
    const int*   dst    = (const int*)d_in[5];
    const float* gin_W  = (const float*)d_in[6];
    const float* gin_b  = (const float*)d_in[7];
    const float* ndW    = (const float*)d_in[8];
    const float* nd_b   = (const float*)d_in[9];
    const float* edW    = (const float*)d_in[10];
    const float* ed_b   = (const float*)d_in[11];
    const float* nffnW  = (const float*)d_in[12];
    const float* nffn_b = (const float*)d_in[13];
    const float* effnW  = (const float*)d_in[14];
    const float* effn_b = (const float*)d_in[15];
    const float* npW    = (const float*)d_in[16];
    const float* npb    = (const float*)d_in[17];
    const float* epW    = (const float*)d_in[18];
    const float* epb    = (const float*)d_in[19];

    float* out = (float*)d_out;

    // workspace layout
    unsigned char* ws = (unsigned char*)d_ws;
    size_t off = 0;
    float* aggr = (float*)(ws + off);            off += (size_t)N * 128 * 4;
    __bf16* xg    = (__bf16*)(ws + off); off += (size_t)N * 128 * 2;
    __bf16* ginP  = (__bf16*)(ws + off); off += 128 * 128 * 2;
    __bf16* ndP   = (__bf16*)(ws + off); off += 32 * 32 * 2;
    __bf16* edP   = (__bf16*)(ws + off); off += 32 * 32 * 2;
    __bf16* nffnP = (__bf16*)(ws + off); off += 160 * 128 * 2;
    __bf16* effnP = (__bf16*)(ws + off); off += 288 * 128 * 2;

    // 0) zero the scatter accumulator (must be re-done every call)
    {
        size_t n = (size_t)N * 128;
        k_fill_zero<<<(unsigned)((n + 255) / 256), 256, 0, stream>>>(aggr, n);
    }
    // 1) pack weights into bf16 B-fragment layout
    k_pack_w<<<(128 * 128 + 255) / 256, 256, 0, stream>>>(gin_W, 128, 128, ginP);
    k_pack_w<<<(32 * 32 + 255) / 256, 256, 0, stream>>>(ndW, 32, 32, ndP);
    k_pack_w<<<(32 * 32 + 255) / 256, 256, 0, stream>>>(edW, 32, 32, edP);
    k_pack_w<<<(160 * 128 + 255) / 256, 256, 0, stream>>>(nffnW, 160, 128, nffnP);
    k_pack_w<<<(288 * 128 + 255) / 256, 256, 0, stream>>>(effnW, 288, 128, effnP);

    // 2) message + scatter-add (8 waves / block, one wave per edge)
    k_scatter<<<(E + 7) / 8, 256, 0, stream>>>(nfeat, efeat, src, dst, aggr, E);

    // 3) node pipeline: 4 waves / block, 16 rows / wave
    {
        int tiles = (N + 15) / 16;
        k_node<<<(tiles + 3) / 4, 128, 0, stream>>>(
            nfeat, aggr, ndist, ginP, gin_b, ndP, nd_b, nffnP, nffn_b,
            npW, npb, xg, out, N);
    }
    // 4) edge pipeline: 4 waves / block, 32 edges / wave
    {
        int tiles = (E + 31) / 32;
        k_edge<<<(tiles + 3) / 4, 128, 0, stream>>>(
            xg, edist, src, dst, edP, ed_b, effnP, effn_b,
            epW, epb, out, E, N);
    }
    (void)n_in; (void)out_size; (void)ws_size;
}
